// MLNN_3968549782347
// MI455X (gfx1250) — matrix operand order repaired and checked by reading, compiled, not run on hardware
//
#include <hip/hip_runtime.h>

// ---------------------------------------------------------------------------
// Problem dims (from reference)
// ---------------------------------------------------------------------------
#define B_DIM   2048
#define DIN     1024
#define H_DIM   2048
#define DOUT    1024
#define NEXP    8
#define NROUTED 4
#define BN_EPS  1e-5f

// GEMM tiling
#define BM   128
#define BN_T 64
#define BK   32
#define LDSS 40   // padded LDS row stride in bf16 elements (80B = 64B row + 16B pad)

typedef __attribute__((ext_vector_type(16))) __bf16       v16bf;
typedef __attribute__((ext_vector_type(8)))  float        v8f;
typedef __attribute__((ext_vector_type(4)))  int          v4i;
typedef __attribute__((ext_vector_type(8)))  int          v8i;
typedef __attribute__((ext_vector_type(4)))  unsigned int v4u;

union Frag {
    uint4 u[2];   // 32 bytes
    v16bf v;      // 16 x bf16
};

__device__ __forceinline__ unsigned int pack_bf16(float a, float b) {
    unsigned int ua = __float_as_uint(a);
    unsigned int ub = __float_as_uint(b);
    ua = (ua + 0x7FFFu + ((ua >> 16) & 1u)) >> 16;   // RNE truncation
    ub = (ub + 0x7FFFu + ((ub >> 16) & 1u)) >> 16;
    return (ua & 0xFFFFu) | (ub << 16);
}

// ---------------------------------------------------------------------------
// CDNA5 async global->LDS copy (ASYNCcnt-tracked)
// ---------------------------------------------------------------------------
__device__ __forceinline__ void async_b128(const void* g, void* lds_generic) {
#if __has_builtin(__builtin_amdgcn_global_load_async_to_lds_b128)
    __builtin_amdgcn_global_load_async_to_lds_b128(
        (v4i*)g, (v4i*)lds_generic, 0, 0);
#else
    unsigned lo = (unsigned)(unsigned long long)
        (__attribute__((address_space(3))) void*)lds_generic;
    asm volatile("global_load_async_to_lds_b128 %0, %1, off"
                 :
                 : "v"(lo), "v"(g)
                 : "memory");
#endif
}

__device__ __forceinline__ void wait_async0() {
#if __has_builtin(__builtin_amdgcn_s_wait_asynccnt)
    __builtin_amdgcn_s_wait_asynccnt(0);
#else
    asm volatile("s_wait_asynccnt 0x0" ::: "memory");
#endif
}
__device__ __forceinline__ void wait_async2() {
#if __has_builtin(__builtin_amdgcn_s_wait_asynccnt)
    __builtin_amdgcn_s_wait_asynccnt(2);
#else
    asm volatile("s_wait_asynccnt 0x2" ::: "memory");
#endif
}

// ---------------------------------------------------------------------------
// CDNA5 Tensor Data Mover: one 128x32(bf16) 2D tile -> LDS, hardware-padded
// rows (pad_interval = 16 DWORDs data, pad_amount = 4 DWORDs) to match LDSS.
// D# built per CDNA5 ISA (data_size=1 -> 2-byte units).
// Toolchain uses the 6-arg builtin: (v4u g0, v8i g1, v4i g2, v4i g3, v8i, cpol)
// ---------------------------------------------------------------------------
#if __has_builtin(__builtin_amdgcn_tensor_load_to_lds)
#define HAVE_TDM 1
#else
#define HAVE_TDM 0
#endif

__device__ __forceinline__ void tdm_load_tile(const void* gaddr, unsigned lds_off,
                                              int lda_elems) {
#if HAVE_TDM
    const unsigned long long ga = (unsigned long long)gaddr;
    v4u g0;
    g0.x = 1u;                                        // count=1, user descriptor
    g0.y = lds_off;                                   // lds_addr (bytes)
    g0.z = (unsigned)ga;                              // global_addr[31:0]
    g0.w = (unsigned)(ga >> 32) | (2u << 30);         // global_addr[56:32] | type=2
    v8i g1;
    g1.s0 = (int)((1u << 16)      // data_size = 1 (2 bytes)
                | (1u << 20)      // pad_enable
                | (3u << 22)      // pad_interval: 16 DWORDs (64B row)
                | (3u << 25));    // pad_amount:    4 DWORDs (16B pad)
    g1.s1 = (int)(32u << 16);     // tensor_dim0[15:0]=32 (valid width)
    g1.s2 = (int)(128u << 16);    // tensor_dim0 hi=0 | tensor_dim1[15:0]=128
    g1.s3 = (int)(32u << 16);     // tensor_dim1 hi=0 | tile_dim0=32
    g1.s4 = 128;                  // tile_dim1=128 | tile_dim2=0
    g1.s5 = lda_elems;            // tensor_dim0_stride[31:0]
    g1.s6 = 0;                    // stride hi | tensor_dim1_stride lo
    g1.s7 = 0;
    v4i z4 = {0, 0, 0, 0};
    v8i z8 = {0, 0, 0, 0, 0, 0, 0, 0};
    __builtin_amdgcn_tensor_load_to_lds(g0, g1, z4, z4, z8, 0);
#else
    (void)gaddr; (void)lds_off; (void)lda_elems;
#endif
}

__device__ __forceinline__ void wait_tensor0() {
#if __has_builtin(__builtin_amdgcn_s_wait_tensorcnt)
    __builtin_amdgcn_s_wait_tensorcnt((short)0);
#else
    asm volatile("s_wait_tensorcnt 0x0" ::: "memory");
#endif
}
__device__ __forceinline__ void wait_tensor1() {
#if __has_builtin(__builtin_amdgcn_s_wait_tensorcnt)
    __builtin_amdgcn_s_wait_tensorcnt((short)1);
#else
    asm volatile("s_wait_tensorcnt 0x1" ::: "memory");
#endif
}

// ---------------------------------------------------------------------------
// Tiled GEMM: Out[m,n] = relu( A @ W + bias ), bf16 WMMA, f32 accum.
//   A: bf16 [*, lda]. Contiguous rows (perm==null): TDM tensor_load_to_lds.
//      Gathered rows (perm!=null): per-lane global_load_async_to_lds_b128.
//   W: fp32 [K, ldw] row-major, converted bf16 + transposed while staging.
//   blockIdx.z = expert segment when segCount != nullptr
// ---------------------------------------------------------------------------
__global__ __launch_bounds__(256)
void gemm_bf16_relu(const unsigned short* __restrict__ Abf, int lda,
                    const float* __restrict__ W, int ldw, long wzstride,
                    const float* __restrict__ bias, int bzstride,
                    float* __restrict__ Out, int ldo,
                    int M, int N, int K,
                    const int* __restrict__ perm,
                    const int* __restrict__ segStart,
                    const int* __restrict__ segCount)
{
    __shared__ unsigned short sA[2][BM * LDSS];   // double-buffered, 2 x 10 KB
    __shared__ unsigned short sB[BN_T * LDSS];    // 5 KB

    const int tid = threadIdx.x;
    const int z   = blockIdx.z;

    int base = 0, cnt = M;
    if (segCount) { base = segStart[z]; cnt = segCount[z]; }
    const int m0 = blockIdx.y * BM;
    if (m0 >= cnt) return;
    const int n0 = blockIdx.x * BN_T;

    const float* Wz = W    + (long)z * wzstride;
    const float* bz = bias + (long)z * bzstride;

#if HAVE_TDM
    const bool useTdm = (perm == nullptr);        // contiguous full tiles only
#else
    const bool useTdm = false;
#endif
    const int  lane = tid & 31;
    const int  wave = tid >> 5;

    // LDS byte addresses of the two A buffers (for TDM descriptor)
    const unsigned ldsA0 = (unsigned)(unsigned long long)
        (__attribute__((address_space(3))) unsigned short*)&sA[0][0];
    const unsigned ldsA1 = (unsigned)(unsigned long long)
        (__attribute__((address_space(3))) unsigned short*)&sA[1][0];

    // per-lane async A staging (gathered path): thread t copies 32B of row t>>1
    const int ar = tid >> 1;           // 0..127 row in tile
    const int ac = (tid & 1) * 16;     // ushort offset within 32-wide row
    int rowIdx;
    {
        const int sl = m0 + ar;
        const bool ok = sl < cnt;
        rowIdx = ok ? (perm ? perm[base + sl] : sl) : 0;  // pad rows copy row 0
    }
    const unsigned short* gRow = Abf + (size_t)rowIdx * lda;

    // B staging: thread t packs k-pair (2*(t>>4), +1) for 4 columns
    const int bk2 = (tid >> 4) * 2;    // 0,2,..,30
    const int bnv = (tid & 15) * 4;    // n within tile

    const int wm   = wave & 3;         // 4 waves along M (wm*32)
    const int wn   = wave >> 2;        // 2 waves along N (wn*32)
    const int half = lane >> 4;
    const int l16  = lane & 15;

    v8f acc[2][2] = {};

    // prologue: first A tile into buffer 0
    if (useTdm) {
        if (wave == 0)
            tdm_load_tile(Abf + (size_t)m0 * lda, ldsA0, lda);
    } else {
        unsigned short* dst = &sA[0][ar * LDSS + ac];
        async_b128(gRow + ac,     dst);
        async_b128(gRow + ac + 8, dst + 8);
    }
    int buf = 0;

    for (int k0 = 0; k0 < K; k0 += BK) {
        // ---- stage B: fp32 row-major [32 x 64] -> bf16 LDS column-major ----
        {
            const float4 w0 = *(const float4*)(Wz + (long)(k0 + bk2)     * ldw + n0 + bnv);
            const float4 w1 = *(const float4*)(Wz + (long)(k0 + bk2 + 1) * ldw + n0 + bnv);
            *(unsigned int*)&sB[(bnv + 0) * LDSS + bk2] = pack_bf16(w0.x, w1.x);
            *(unsigned int*)&sB[(bnv + 1) * LDSS + bk2] = pack_bf16(w0.y, w1.y);
            *(unsigned int*)&sB[(bnv + 2) * LDSS + bk2] = pack_bf16(w0.z, w1.z);
            *(unsigned int*)&sB[(bnv + 3) * LDSS + bk2] = pack_bf16(w0.w, w1.w);
        }
        const bool more = (k0 + BK) < K;
        if (more)   // prefetch next K-chunk of weights
            __builtin_prefetch(Wz + (long)(k0 + BK + bk2) * ldw + n0 + bnv, 0, 0);

        // ---- issue next A tile, wait for the current one, then barrier ----
        if (useTdm) {
            if (wave == 0) {
                if (more) {
                    tdm_load_tile(Abf + (size_t)m0 * lda + (k0 + BK),
                                  (buf ? ldsA0 : ldsA1), lda);
                    wait_tensor1();   // 1 outstanding = next tile; current landed
                } else {
                    wait_tensor0();
                }
            }
        } else {
            if (more) {
                unsigned short* dst = &sA[buf ^ 1][ar * LDSS + ac];
                const unsigned short* src = gRow + (k0 + BK);
                async_b128(src + ac,     dst);
                async_b128(src + ac + 8, dst + 8);
                wait_async2();        // 2 outstanding = next tile; current landed
            } else {
                wait_async0();
            }
        }
        __syncthreads();

        // ---- load fragments per ISA layout, 4 WMMAs per wave ----
        Frag a0, a1, b0, b1;
        {
            const unsigned short* pa = &sA[buf][(wm * 32 + l16) * LDSS + half * 8];
            a0.u[0] = *(const uint4*)(pa);
            a0.u[1] = *(const uint4*)(pa + 16);
            pa += 16 * LDSS;
            a1.u[0] = *(const uint4*)(pa);
            a1.u[1] = *(const uint4*)(pa + 16);
        }
        {
            const unsigned short* pb = &sB[(wn * 32 + l16) * LDSS + half * 8];
            b0.u[0] = *(const uint4*)(pb);
            b0.u[1] = *(const uint4*)(pb + 16);
            pb += 16 * LDSS;
            b1.u[0] = *(const uint4*)(pb);
            b1.u[1] = *(const uint4*)(pb + 16);
        }
        acc[0][0] = __builtin_amdgcn_wmma_f32_16x16x32_bf16(false, a0.v, false, b0.v,
                                                            (short)0, acc[0][0], false, false);
        acc[0][1] = __builtin_amdgcn_wmma_f32_16x16x32_bf16(false, a0.v, false, b1.v,
                                                            (short)0, acc[0][1], false, false);
        acc[1][0] = __builtin_amdgcn_wmma_f32_16x16x32_bf16(false, a1.v, false, b0.v,
                                                            (short)0, acc[1][0], false, false);
        acc[1][1] = __builtin_amdgcn_wmma_f32_16x16x32_bf16(false, a1.v, false, b1.v,
                                                            (short)0, acc[1][1], false, false);
        __syncthreads();
        buf ^= 1;
    }

    // ---- epilogue: bias + relu, scatter rows back by perm ----
    const float bv0 = bz[n0 + wn * 32 + l16];
    const float bv1 = bz[n0 + wn * 32 + 16 + l16];
    #pragma unroll
    for (int tm = 0; tm < 2; ++tm) {
        #pragma unroll
        for (int v = 0; v < 8; ++v) {
            const int sl = m0 + wm * 32 + tm * 16 + half * 8 + v;  // C layout: VGPR v -> M=half*8+v
            if (sl >= cnt) continue;
            const int orow = perm ? perm[base + sl] : sl;
            float* op = Out + (long)orow * ldo + n0 + wn * 32 + l16;
            float r0 = acc[tm][0][v] + bv0;
            float r1 = acc[tm][1][v] + bv1;
            op[0]  = r0 > 0.f ? r0 : 0.f;
            op[16] = r1 > 0.f ? r1 : 0.f;
        }
    }
}

// ---------------------------------------------------------------------------
// fp32 -> bf16 bulk convert (for input x)
// ---------------------------------------------------------------------------
__global__ __launch_bounds__(256)
void to_bf16(const float* __restrict__ src, unsigned short* __restrict__ dst, int n)
{
    const int i = (blockIdx.x * 256 + threadIdx.x) * 4;
    if (i >= n) return;
    const float4 v = *(const float4*)(src + i);
    uint2 pk;
    pk.x = pack_bf16(v.x, v.y);
    pk.y = pack_bf16(v.z, v.w);
    *(uint2*)(dst + i) = pk;
}

// ---------------------------------------------------------------------------
// BatchNorm stats -> fused scale/shift (deterministic tree reduction)
// ---------------------------------------------------------------------------
__global__ __launch_bounds__(256)
void bn_stats(const float* __restrict__ h,
              const float* __restrict__ gamma, const float* __restrict__ beta,
              float* __restrict__ scale, float* __restrict__ shift,
              int Bn, int Hn)
{
    __shared__ float rs[256];
    __shared__ float rq[256];
    const int cl  = threadIdx.x & 15;
    const int rl  = threadIdx.x >> 4;
    const int col = blockIdx.x * 16 + cl;
    float s = 0.f, q = 0.f;
    for (int r = rl; r < Bn; r += 16) {
        const float x = h[(long)r * Hn + col];
        s += x;
        q += x * x;
    }
    rs[threadIdx.x] = s;
    rq[threadIdx.x] = q;
    __syncthreads();
    for (int st = 128; st >= 16; st >>= 1) {
        if (threadIdx.x < st) {
            rs[threadIdx.x] += rs[threadIdx.x + st];
            rq[threadIdx.x] += rq[threadIdx.x + st];
        }
        __syncthreads();
    }
    if (threadIdx.x < 16) {
        const float mean = rs[threadIdx.x] / (float)Bn;
        const float var  = rq[threadIdx.x] / (float)Bn - mean * mean;
        const float rstd = rsqrtf(var + BN_EPS);
        const float sc   = gamma[col] * rstd;
        scale[col] = sc;
        shift[col] = beta[col] - mean * sc;
    }
}

// ---------------------------------------------------------------------------
// Apply BN: write fp32 copy (router) AND bf16 copy (GEMM A operand).
// ---------------------------------------------------------------------------
__global__ __launch_bounds__(256)
void bn_apply(const float* __restrict__ h,
              const float* __restrict__ scale, const float* __restrict__ shift,
              float* __restrict__ out32, unsigned short* __restrict__ outbf, int Hn)
{
    const long i = ((long)blockIdx.x * 256 + threadIdx.x) * 4;
    const int  c = (int)(i & (Hn - 1));   // Hn is a power of two
    const float4 x  = *(const float4*)(h + i);
    const float4 sc = *(const float4*)(scale + c);
    const float4 sh = *(const float4*)(shift + c);
    float4 y;
    y.x = fmaf(x.x, sc.x, sh.x);
    y.y = fmaf(x.y, sc.y, sh.y);
    y.z = fmaf(x.z, sc.z, sh.z);
    y.w = fmaf(x.w, sc.w, sh.w);
    *(float4*)(out32 + i) = y;
    uint2 pk;
    pk.x = pack_bf16(y.x, y.y);
    pk.y = pack_bf16(y.z, y.w);
    *(uint2*)(outbf + i) = pk;
}

// ---------------------------------------------------------------------------
// Router: argmax_e( hN @ dqn_W + dqn_b ). One wave32 per row.
// ---------------------------------------------------------------------------
__global__ __launch_bounds__(256)
void router_argmax(const float* __restrict__ hN,
                   const float* __restrict__ dqnW, const float* __restrict__ dqnb,
                   int* __restrict__ actions, int Hn)
{
    const int lane = threadIdx.x & 31;
    const int wave = threadIdx.x >> 5;
    const int row  = blockIdx.x * 8 + wave;
    const float* hr = hN + (long)row * Hn;
    float acc[NEXP] = {};
    for (int k = lane; k < Hn; k += 32) {
        const float a  = hr[k];
        const float4 w0 = *(const float4*)(dqnW + k * NEXP);
        const float4 w1 = *(const float4*)(dqnW + k * NEXP + 4);
        acc[0] = fmaf(a, w0.x, acc[0]);
        acc[1] = fmaf(a, w0.y, acc[1]);
        acc[2] = fmaf(a, w0.z, acc[2]);
        acc[3] = fmaf(a, w0.w, acc[3]);
        acc[4] = fmaf(a, w1.x, acc[4]);
        acc[5] = fmaf(a, w1.y, acc[5]);
        acc[6] = fmaf(a, w1.z, acc[6]);
        acc[7] = fmaf(a, w1.w, acc[7]);
    }
    #pragma unroll
    for (int off = 16; off > 0; off >>= 1)
        #pragma unroll
        for (int e = 0; e < NEXP; ++e)
            acc[e] += __shfl_down(acc[e], off);
    if (lane == 0) {
        int   best = 0;
        float bv   = acc[0] + dqnb[0];
        #pragma unroll
        for (int e = 1; e < NEXP; ++e) {
            const float v = acc[e] + dqnb[e];
            if (v > bv) { bv = v; best = e; }   // strict >: first max, matches jnp.argmax
        }
        actions[row] = best;
    }
}

// ---------------------------------------------------------------------------
// Stable partition of rows by expert (deterministic, serial; trivial cost).
// ---------------------------------------------------------------------------
__global__ void build_perm(const int* __restrict__ actions, int Bn,
                           int* __restrict__ perm, int* __restrict__ segStart,
                           int* __restrict__ segCount)
{
    if (threadIdx.x != 0 || blockIdx.x != 0) return;
    int cnt[NEXP];
    for (int e = 0; e < NEXP; ++e) cnt[e] = 0;
    for (int r = 0; r < Bn; ++r) cnt[actions[r]]++;
    int off[NEXP];
    int s = 0;
    for (int e = 0; e < NEXP; ++e) {
        segStart[e] = s;
        segCount[e] = cnt[e];
        off[e] = s;
        s += cnt[e];
    }
    for (int r = 0; r < Bn; ++r) {
        const int e = actions[r];
        perm[off[e]++] = r;
    }
}

// ---------------------------------------------------------------------------
// Host-side orchestration (graph-capture safe: only kernel launches on stream)
// ---------------------------------------------------------------------------
extern "C" void kernel_launch(void* const* d_in, const int* in_sizes, int n_in,
                              void* d_out, int out_size, void* d_ws, size_t ws_size,
                              hipStream_t stream)
{
    const float* x           = (const float*)d_in[0];
    const float* start_W     = (const float*)d_in[1];
    const float* start_b     = (const float*)d_in[2];
    const float* start_gamma = (const float*)d_in[3];
    const float* start_beta  = (const float*)d_in[4];
    const float* expert_W    = (const float*)d_in[5];
    const float* expert_b    = (const float*)d_in[6];
    const float* bn_gamma    = (const float*)d_in[7];
    const float* bn_beta     = (const float*)d_in[8];
    const float* dqn_W       = (const float*)d_in[9];
    const float* dqn_b       = (const float*)d_in[10];
    const float* end_W       = (const float*)d_in[11];
    const float* end_b       = (const float*)d_in[12];
    float* out = (float*)d_out;

    // workspace carve-out (~60 MB)
    float*          hA       = (float*)d_ws;
    float*          hB       = hA + (size_t)B_DIM * H_DIM;
    float*          hN32     = hB + (size_t)B_DIM * H_DIM;
    unsigned short* hNbf     = (unsigned short*)(hN32 + (size_t)B_DIM * H_DIM);
    unsigned short* xbf      = hNbf + (size_t)B_DIM * H_DIM;
    float*          scale    = (float*)(xbf + (size_t)B_DIM * DIN);
    float*          shift    = scale + H_DIM;
    int*            actions  = (int*)(shift + H_DIM);
    int*            perm     = actions + B_DIM;
    int*            segStart = perm + B_DIM;
    int*            segCount = segStart + NEXP;

    const dim3 blk(256);

    // convert x to bf16 once (A operand of start GEMM)
    to_bf16<<<dim3((B_DIM * DIN) / 1024), blk, 0, stream>>>(x, xbf, B_DIM * DIN);

    // start: h = relu(x @ start_W + b)   (contiguous A -> TDM path)
    gemm_bf16_relu<<<dim3(H_DIM / BN_T, B_DIM / BM, 1), blk, 0, stream>>>(
        xbf, DIN, start_W, H_DIM, 0L, start_b, 0,
        hA, H_DIM, B_DIM, H_DIM, DIN, nullptr, nullptr, nullptr);
    bn_stats<<<dim3(H_DIM / 16), blk, 0, stream>>>(
        hA, start_gamma, start_beta, scale, shift, B_DIM, H_DIM);

    for (int i = 0; i < NROUTED; ++i) {
        bn_apply<<<dim3((B_DIM * H_DIM) / 1024), blk, 0, stream>>>(
            hA, scale, shift, hN32, hNbf, H_DIM);
        router_argmax<<<dim3(B_DIM / 8), blk, 0, stream>>>(
            hN32, dqn_W, dqn_b, actions, H_DIM);
        build_perm<<<1, 32, 0, stream>>>(actions, B_DIM, perm, segStart, segCount);
        // grouped expert GEMM: gathered A rows -> per-lane async-to-LDS path
        gemm_bf16_relu<<<dim3(H_DIM / BN_T, B_DIM / BM, NEXP), blk, 0, stream>>>(
            hNbf, H_DIM,
            expert_W + (size_t)i * NEXP * H_DIM * H_DIM, H_DIM, (long)H_DIM * H_DIM,
            expert_b + (size_t)i * NEXP * H_DIM, H_DIM,
            hB, H_DIM, B_DIM, H_DIM, H_DIM, perm, segStart, segCount);
        bn_stats<<<dim3(H_DIM / 16), blk, 0, stream>>>(
            hB, bn_gamma + i * H_DIM, bn_beta + i * H_DIM, scale, shift, B_DIM, H_DIM);
        float* t = hA; hA = hB; hB = t;
    }

    // end: out = relu(BN(h) @ end_W + b)   (contiguous A -> TDM path)
    bn_apply<<<dim3((B_DIM * H_DIM) / 1024), blk, 0, stream>>>(
        hA, scale, shift, hN32, hNbf, H_DIM);
    gemm_bf16_relu<<<dim3(DOUT / BN_T, B_DIM / BM, 1), blk, 0, stream>>>(
        hNbf, H_DIM, end_W, DOUT, 0L, end_b, 0,
        out, DOUT, B_DIM, H_DIM, DOUT, nullptr, nullptr, nullptr);
}
